// Decoder_33449205302010
// MI455X (gfx1250) — matrix operand order repaired and checked by reading, not compiled, not run on hardware
//
#include <hip/hip_runtime.h>
#include <hip/hip_bf16.h>

#define H     256
#define NMEL  80
#define DENC  512
#define NB    64
#define SS    256
#define TT    768
#define TD    767
#define NWG   8

typedef __bf16        v16bf __attribute__((ext_vector_type(16)));
typedef float         v8f   __attribute__((ext_vector_type(8)));
typedef unsigned int  v4u   __attribute__((ext_vector_type(4)));

union Frag { v4u q[2]; v16bf v; };

__device__ __forceinline__ unsigned short f2bf(float f) {
  unsigned int u = __float_as_uint(f);
  unsigned int r = u + 0x7FFFu + ((u >> 16) & 1u);
  return (unsigned short)(r >> 16);
}
__device__ __forceinline__ float sigm(float x) { return 1.0f / (1.0f + __expf(-x)); }

// ---------------- utility kernels ----------------

__global__ void zero_u32_kernel(unsigned int* p, long long n) {
  long long i = (long long)blockIdx.x * blockDim.x + threadIdx.x;
  if (i < n) p[i] = 0u;
}

// dst bf16 [1024][Kpad] = [Whh | Wih | zero-pad] per scan
__global__ void build_weight_kernel(unsigned short* dst, const float* whh,
                                    const float* wih, int Din, int Kpad) {
  long long i = (long long)blockIdx.x * blockDim.x + threadIdx.x;
  long long n = 1024LL * Kpad;
  if (i >= n) return;
  int row = (int)(i / Kpad), k = (int)(i % Kpad);
  float v = 0.0f;
  if (k < H)                v = whh[(long long)row * H + k];
  else if ((k - H) < Din)   v = wih[(long long)row * Din + (k - H)];
  dst[i] = f2bf(v);
}

__global__ void conv_bf16_kernel(unsigned short* dst, const float* src, long long n) {
  long long i = (long long)blockIdx.x * blockDim.x + threadIdx.x;
  if (i < n) dst[i] = f2bf(src[i]);
}

// encoder x: dst [S][N][512] bf16 from enc_out [N][S][512] f32
__global__ void build_encx_kernel(unsigned short* dst, const float* enc) {
  long long i = (long long)blockIdx.x * blockDim.x + threadIdx.x;
  long long n = (long long)SS * NB * DENC;
  if (i >= n) return;
  int s  = (int)(i / (NB * DENC));
  int rm = (int)(i % (NB * DENC));
  int nn = rm / DENC, k = rm % DENC;
  dst[i] = f2bf(enc[(long long)nn * SS * DENC + (long long)s * DENC + k]);
}

// teacher: dst [TD][N][96] bf16; t==0 -> init_ctx, t>=1 -> mels[n][k][t-1]; k>=80 -> 0
__global__ void build_teacher_kernel(unsigned short* dst, const float* mels, const float* ictx) {
  long long i = (long long)blockIdx.x * blockDim.x + threadIdx.x;
  long long n = (long long)TD * NB * 96;
  if (i >= n) return;
  int t  = (int)(i / (NB * 96));
  int rm = (int)(i % (NB * 96));
  int nn = rm / 96, k = rm % 96;
  float v = 0.0f;
  if (k < NMEL)
    v = (t == 0) ? ictx[k] : mels[(long long)nn * NMEL * TT + (long long)k * TT + (t - 1)];
  dst[i] = f2bf(v);
}

// ---------------- grid barrier (8 persistent WGs) ----------------

__device__ __forceinline__ void grid_barrier(unsigned int* bar) {
  __syncthreads();
  __threadfence();                              // release: flush h stores to device scope
  if (threadIdx.x == 0) {
    volatile unsigned int* vgen = bar + 1;
    unsigned int gen = *vgen;
    if (atomicAdd(bar, 1u) == (unsigned)(NWG - 1)) {
      *(volatile unsigned int*)bar = 0u;
      __threadfence();
      atomicAdd((unsigned int*)(bar + 1), 1u);  // open barrier
    } else {
      while (*vgen == gen) { __builtin_amdgcn_s_sleep(1); }
    }
  }
  __syncthreads();
  __threadfence();                              // acquire: invalidate stale h in WGP cache
}

// ---------------- persistent LSTM scan ----------------
// 8 WGs x 256 threads. WG w owns h-cols [32w,32w+32); computes gate cols
// {g*256 + 32w .. +32 | g=0..3}. C kept in registers. bf16 WMMA, f32 accum.
__global__ void __launch_bounds__(256) lstm_scan_kernel(
    const unsigned short* __restrict__ W,     // [1024][Kpad] bf16 (row = gate col)
    const float*          __restrict__ bias,  // [1024] f32
    const unsigned short* __restrict__ xbuf,  // [steps][NB][xrow] bf16
    int xrow, int Kpad, int steps, int rev,
    const unsigned short* __restrict__ h0,    // [NB][H] bf16
    const float*          __restrict__ c0,    // [NB][H] f32 or null
    unsigned short*       __restrict__ hbuf,  // [2][NB][H] bf16 ping-pong
    unsigned short*       __restrict__ ybuf,  // null ok
    int yrow, int ycol0,
    unsigned short*       __restrict__ hfin,  // null ok
    float*                __restrict__ cfin,  // null ok
    unsigned int*         __restrict__ bar) {
  __shared__ float gl[8][64][16];             // gate pre-activations, 32 KB

  const int tid  = threadIdx.x;
  const int wv   = tid >> 5, lane = tid & 31;
  const int hi   = lane >> 4, lm = lane & 15;
  const int wg   = blockIdx.x;
  const int gate = wv >> 1, sub = wv & 1;
  const int ncol = gate * H + wg * 32 + sub * 16 + lm;   // W row this lane supplies
  const unsigned short* brow0 = W + (long long)ncol * Kpad;

  // per-thread persistent cell state + biases (8 elems: row=idx>>5, col=idx&31)
  float cr[8], bi[8], bf_[8], bg[8], bo[8];
  for (int e = 0; e < 8; ++e) {
    int idx = tid * 8 + e;
    int row = idx >> 5, col = (idx & 31) + wg * 32;
    cr[e]  = c0 ? c0[row * H + col] : 0.0f;
    bi[e]  = bias[0 * H + col];
    bf_[e] = bias[1 * H + col];
    bg[e]  = bias[2 * H + col];
    bo[e]  = bias[3 * H + col];
  }

  for (int t = 0; t < steps; ++t) {
    const int mem = rev ? (steps - 1 - t) : t;
    const unsigned short* hsrc  = (t == 0) ? h0 : (hbuf + (size_t)((t - 1) & 1) * NB * H);
    const unsigned short* xstep = xbuf + (size_t)mem * NB * xrow;

    if (t + 1 < steps) {  // prefetch next step's x slice (global_prefetch_b8)
      int mem2 = rev ? (steps - 2 - t) : (t + 1);
      __builtin_prefetch(xbuf + (size_t)mem2 * NB * xrow + (size_t)tid * ((NB * xrow) / 256), 0, 3);
    }

    v8f acc[4];
    for (int mt = 0; mt < 4; ++mt)
      for (int r = 0; r < 8; ++r) acc[mt][r] = 0.0f;

    for (int k0 = 0; k0 < Kpad; k0 += 32) {
      Frag b;  // B 32x16: lanes 0-15 K=k0..k0+15, lanes 16-31 K=k0+16..k0+31 (contig)
      b.q[0] = *(const v4u*)(brow0 + k0 + hi * 16);
      b.q[1] = *(const v4u*)(brow0 + k0 + hi * 16 + 8);
      for (int mt = 0; mt < 4; ++mt) {
        int m = mt * 16 + lm;
        const unsigned short* arow;
        int kk;
        if (k0 < H) { arow = hsrc + (long long)m * H;     kk = k0; }
        else        { arow = xstep + (long long)m * xrow; kk = k0 - H; }
        Frag a;  // A 16x32: lanes 0-15 K=kk..kk+15, lanes 16-31 K=kk+16..kk+31 (contig)
        a.q[0] = *(const v4u*)(arow + kk + hi * 16);
        a.q[1] = *(const v4u*)(arow + kk + hi * 16 + 8);
        acc[mt] = __builtin_amdgcn_wmma_f32_16x16x32_bf16(
            false, a.v, false, b.v, (short)0, acc[mt], false, false);
      }
    }

    // C layout: lane N=lm, VGPR r -> M = r + 8*hi within tile
    for (int mt = 0; mt < 4; ++mt)
      for (int r = 0; r < 8; ++r)
        gl[wv][mt * 16 + hi * 8 + r][lm] = acc[mt][r];
    __syncthreads();

    unsigned short* hdst = hbuf + (size_t)(t & 1) * NB * H;
    for (int e = 0; e < 8; ++e) {
      int idx = tid * 8 + e;
      int row = idx >> 5, col = idx & 31;
      int sb = col >> 4, cc = col & 15;
      float pi = gl[0 + sb][row][cc] + bi[e];
      float pf = gl[2 + sb][row][cc] + bf_[e];
      float pg = gl[4 + sb][row][cc] + bg[e];
      float po = gl[6 + sb][row][cc] + bo[e];
      float c  = sigm(pf) * cr[e] + sigm(pi) * tanhf(pg);
      cr[e] = c;
      float h = sigm(po) * tanhf(c);
      unsigned short hb = f2bf(h);
      int gcol = wg * 32 + col;
      hdst[row * H + gcol] = hb;
      if (ybuf) ybuf[(size_t)mem * NB * yrow + (long long)row * yrow + ycol0 + gcol] = hb;
      if (t == steps - 1) {
        if (hfin) hfin[row * H + gcol] = hb;
        if (cfin) cfin[row * H + gcol] = c;
      }
    }
    grid_barrier(bar);
  }
}

// ---------------- FC: out[n][c][t] = d3 @ fcW^T + fcb ----------------
// rows r = t*64+n (49088 = 3068 M-tiles), K=256, 80 cols = 5 N-tiles
__global__ void __launch_bounds__(32) fc_kernel(const unsigned short* __restrict__ d3,
                                                const unsigned short* __restrict__ fcw,
                                                const float* __restrict__ fcb,
                                                float* __restrict__ out) {
  const int mt = blockIdx.x;
  const int lane = threadIdx.x, hi = lane >> 4, lm = lane & 15;
  v8f acc[5];
  for (int nt = 0; nt < 5; ++nt)
    for (int r = 0; r < 8; ++r) acc[nt][r] = 0.0f;

  const unsigned short* arow = d3 + (long long)(mt * 16 + lm) * H;
  for (int k0 = 0; k0 < H; k0 += 32) {
    Frag a;
    a.q[0] = *(const v4u*)(arow + k0 + hi * 16);
    a.q[1] = *(const v4u*)(arow + k0 + hi * 16 + 8);
    for (int nt = 0; nt < 5; ++nt) {
      const unsigned short* brow = fcw + (long long)(nt * 16 + lm) * H + k0 + hi * 16;
      Frag b;
      b.q[0] = *(const v4u*)brow;
      b.q[1] = *(const v4u*)(brow + 8);
      acc[nt] = __builtin_amdgcn_wmma_f32_16x16x32_bf16(
          false, a.v, false, b.v, (short)0, acc[nt], false, false);
    }
  }
  for (int nt = 0; nt < 5; ++nt) {
    int col = nt * 16 + lm;
    float bb = fcb[col];
    for (int r = 0; r < 8; ++r) {
      int orow = mt * 16 + hi * 8 + r;
      int t = orow >> 6, nn = orow & 63;
      out[(long long)nn * NMEL * TT + (long long)col * TT + t] = acc[nt][r] + bb;
    }
  }
}

// ---------------- host ----------------

extern "C" void kernel_launch(void* const* d_in, const int* in_sizes, int n_in,
                              void* d_out, int out_size, void* d_ws, size_t ws_size,
                              hipStream_t stream) {
  (void)in_sizes; (void)n_in; (void)ws_size;
  const float* enc_out  = (const float*)d_in[0];
  const float* mels     = (const float*)d_in[1];
  const float* enc_Wih  = (const float*)d_in[4];
  const float* enc_Whh  = (const float*)d_in[5];
  const float* enc_b    = (const float*)d_in[6];
  const float* dec_Wih0 = (const float*)d_in[7];
  const float* dec_Wih  = (const float*)d_in[8];
  const float* dec_Whh  = (const float*)d_in[9];
  const float* dec_b    = (const float*)d_in[10];
  const float* fc_W     = (const float*)d_in[11];
  const float* fc_b     = (const float*)d_in[12];
  const float* init_ctx = (const float*)d_in[13];

  char* ws = (char*)d_ws;
  size_t off = 0;
  auto alloc = [&](size_t bytes) -> void* {
    void* p = ws + off;
    off += (bytes + 255) & ~(size_t)255;
    return p;
  };

  unsigned int*   bar   = (unsigned int*)  alloc(256);
  unsigned short* zeroh = (unsigned short*)alloc((size_t)NB * H * 2);          // bf16 zeros (h0)
  unsigned short* hbuf  = (unsigned short*)alloc((size_t)2 * NB * H * 2);      // h ping-pong
  unsigned short* encW[4];
  for (int s = 0; s < 4; ++s) encW[s] = (unsigned short*)alloc(1024ULL * 768 * 2);
  unsigned short* decW[4];
  decW[0] = (unsigned short*)alloc(1024ULL * 352 * 2);
  for (int l = 1; l < 4; ++l) decW[l] = (unsigned short*)alloc(1024ULL * 512 * 2);
  unsigned short* fcwb  = (unsigned short*)alloc((size_t)NMEL * H * 2);
  unsigned short* encx  = (unsigned short*)alloc((size_t)SS * NB * DENC * 2);
  unsigned short* ency  = (unsigned short*)alloc((size_t)SS * NB * DENC * 2);
  unsigned short* teach = (unsigned short*)alloc((size_t)TD * NB * 96 * 2);
  unsigned short* dbufA = (unsigned short*)alloc((size_t)TD * NB * H * 2);
  unsigned short* dbufB = (unsigned short*)alloc((size_t)TD * NB * H * 2);
  unsigned short* hfin  = (unsigned short*)alloc(4ULL * NB * H * 2);
  float*          cfin  = (float*)         alloc(4ULL * NB * H * 4);

  auto z = [&](void* p, long long n32) {
    zero_u32_kernel<<<dim3((unsigned)((n32 + 255) / 256)), dim3(256), 0, stream>>>(
        (unsigned int*)p, n32);
  };
  z(bar, 64);
  z(zeroh, (long long)NB * H / 2);
  z(d_out, (long long)out_size);

  auto bw = [&](unsigned short* dst, const float* whh, const float* wih, int Din, int Kpad) {
    long long n = 1024LL * Kpad;
    build_weight_kernel<<<dim3((unsigned)((n + 255) / 256)), dim3(256), 0, stream>>>(
        dst, whh, wih, Din, Kpad);
  };
  for (int s = 0; s < 4; ++s)  // s = l*2 + d
    bw(encW[s], enc_Whh + (size_t)s * 1024 * H, enc_Wih + (size_t)s * 1024 * DENC, DENC, 768);
  bw(decW[0], dec_Whh, dec_Wih0, NMEL, 352);
  for (int l = 1; l < 4; ++l)
    bw(decW[l], dec_Whh + (size_t)l * 1024 * H, dec_Wih + (size_t)(l - 1) * 1024 * H, H, 512);

  {
    long long n = (long long)NMEL * H;
    conv_bf16_kernel<<<dim3((unsigned)((n + 255) / 256)), dim3(256), 0, stream>>>(fcwb, fc_W, n);
  }
  {
    long long n = (long long)SS * NB * DENC;
    build_encx_kernel<<<dim3((unsigned)((n + 255) / 256)), dim3(256), 0, stream>>>(encx, enc_out);
  }
  {
    long long n = (long long)TD * NB * 96;
    build_teacher_kernel<<<dim3((unsigned)((n + 255) / 256)), dim3(256), 0, stream>>>(
        teach, mels, init_ctx);
  }

  auto scan = [&](const unsigned short* Wc, const float* bb, const unsigned short* xb,
                  int xrow, int Kpad, int steps, int rev,
                  const unsigned short* h0p, const float* c0p,
                  unsigned short* yb, int yrow, int ycol0,
                  unsigned short* hf, float* cf) {
    lstm_scan_kernel<<<dim3(NWG), dim3(256), 0, stream>>>(
        Wc, bb, xb, xrow, Kpad, steps, rev, h0p, c0p, hbuf, yb, yrow, ycol0, hf, cf, bar);
  };

  // encoder layer 0: fwd -> ency cols [0,256), bwd (reversed in place) -> cols [256,512)
  scan(encW[0], enc_b + 0,    encx, DENC, 768, SS, 0, zeroh, nullptr, ency, DENC, 0,
       hfin + 0 * NB * H, cfin + 0 * NB * H);
  scan(encW[1], enc_b + 1024, encx, DENC, 768, SS, 1, zeroh, nullptr, ency, DENC, H,
       hfin + 1 * NB * H, cfin + 1 * NB * H);
  // encoder layer 1: only final (h,c) needed
  scan(encW[2], enc_b + 2048, ency, DENC, 768, SS, 0, zeroh, nullptr, nullptr, 0, 0,
       hfin + 2 * NB * H, cfin + 2 * NB * H);
  scan(encW[3], enc_b + 3072, ency, DENC, 768, SS, 1, zeroh, nullptr, nullptr, 0, 0,
       hfin + 3 * NB * H, cfin + 3 * NB * H);

  // decoder: layer l inits from encoder scan l's final (h,c)
  scan(decW[0], dec_b + 0,    teach, 96, 352, TD, 0, hfin + 0 * NB * H, cfin + 0 * NB * H,
       dbufA, H, 0, nullptr, nullptr);
  scan(decW[1], dec_b + 1024, dbufA, H,  512, TD, 0, hfin + 1 * NB * H, cfin + 1 * NB * H,
       dbufB, H, 0, nullptr, nullptr);
  scan(decW[2], dec_b + 2048, dbufB, H,  512, TD, 0, hfin + 2 * NB * H, cfin + 2 * NB * H,
       dbufA, H, 0, nullptr, nullptr);
  scan(decW[3], dec_b + 3072, dbufA, H,  512, TD, 0, hfin + 3 * NB * H, cfin + 3 * NB * H,
       dbufB, H, 0, nullptr, nullptr);

  // FC + scatter to [N, NMEL, T] (t = T-1 stays zero from the init pass)
  fc_kernel<<<dim3((TD * NB) / 16), dim3(32), 0, stream>>>(dbufB, fcwb, fc_b, (float*)d_out);
}
